// StaticVarSelection_88038239633910
// MI455X (gfx1250) — matrix operand order repaired and checked by reading, not compiled, not run on hardware
//
#include <hip/hip_runtime.h>
#include <stdint.h>

// ---------------------------------------------------------------------------
// StaticVarSelection for MI455X (gfx1250, wave32, WMMA bf16 16x16x32).
// One wave handles a 16-row batch tile end-to-end. Weights are pre-packed
// into bf16 WMMA B-fragment layout in d_ws (~1.45 MB; assumes ws_size >= 2MB).
// ---------------------------------------------------------------------------

typedef __attribute__((ext_vector_type(16))) __bf16 v16bf;
typedef __attribute__((ext_vector_type(2)))  __bf16 v2bf;
typedef __attribute__((ext_vector_type(8)))  float  v8f;

#define IN_DIM 2048
#define B_ROWS 16384
#define EPSL   1e-5f

// u32-element offsets inside workspace
enum : int {
  OFF_MW1  = 0,        // [2048,64]  -> 64 chunks * 4 tiles * 256 u32
  OFF_MLW  = 65536,    // [2048,32]  -> 64 * 2 * 256
  OFF_MW2  = 98304,    // [64,32]    -> 2 * 2 * 256
  OFF_MGW  = 99328,    // [32,32]    -> 1 * 2 * 256
  OFF_MLW2 = 99840,    // [32,32]
  OFF_W1   = 100352,   // [32][64,64]-> 32 * 2 * 4 * 256
  OFF_W2   = 165888,
  OFF_GW   = 231424,
  OFF_LW   = 296960,
  WS_U32_TOTAL = 362496    // ~1.45 MB
};

// Packed f32x2 -> bf16x2 conversion (prefers v_cvt_pk_bf16_f32).
__device__ __forceinline__ v2bf cvt2(float lo, float hi){
#if __has_builtin(__builtin_amdgcn_cvt_pk_bf16_f32)
  return __builtin_amdgcn_cvt_pk_bf16_f32(lo, hi);
#else
  v2bf r; r[0] = (__bf16)lo; r[1] = (__bf16)hi; return r;
#endif
}
__device__ __forceinline__ __bf16 tobf(float f){ return (__bf16)f; }
__device__ __forceinline__ uint32_t pack2bf(float lo, float hi){
  union { v2bf h; uint32_t u; } x; x.h = cvt2(lo, hi); return x.u;
}

// ---------------------------------------------------------------------------
// Weight pre-pack: W[nmat][Kt][Nt] (f32, row-major) -> bf16 B-fragments.
// Fragment (mat, c, t) holds K=c*32..+31, N=t*16..+15. Inner layout: lane L
// owns column n = t*16+(L&15); VGPR j packs K pair (c*32 + (L>>4)*8 + 2j + (j>=4)*8, +1).
// Stored as u32 frag[32][8] so the consumer does two b128 loads per fragment.
// ---------------------------------------------------------------------------
__global__ void pack_weights_kernel(const float* __restrict__ W, uint32_t* __restrict__ dst,
                                    int Kt, int Nt, int nmat){
  int id = blockIdx.x * blockDim.x + threadIdx.x;
  int chunks = Kt >> 5, tiles = Nt >> 4;
  int total = nmat * chunks * tiles * 256;
  if (id >= total) return;
  int j = id & 7;
  int L = (id >> 3) & 31;
  int rest = id >> 8;
  int t = rest % tiles; rest /= tiles;
  int c = rest % chunks; int mat = rest / chunks;
  int n  = t * 16 + (L & 15);
  int kb = c * 32 + ((L >> 4) << 3) + 2 * j + ((j >> 2) << 3);
  const float* Wm = W + (size_t)mat * Kt * Nt;
  dst[id] = pack2bf(Wm[(size_t)kb * Nt + n], Wm[(size_t)(kb + 1) * Nt + n]);
}

// ---------------------------------------------------------------------------
// Fragment loaders / math helpers
// ---------------------------------------------------------------------------
__device__ __forceinline__ v8f vzero8(){ v8f z = {0.f,0.f,0.f,0.f,0.f,0.f,0.f,0.f}; return z; }

// A fragment (16x32 bf16) from row-major f32 [16 x ld] block, K chunk at kc.
// Lane L owns row (L&15); its 16 K-values live at [kb, kb+8) and [kb+16, kb+24)
// with kb = kc + (L>>4)*8  -> two pairs of b128 loads + packed conversions.
__device__ __forceinline__ v16bf load_a_global(const float* base, int ld, int kc, int lane){
  const float* r = base + (size_t)(lane & 15) * ld + kc + ((lane >> 4) << 3);
  float4 p0 = *(const float4*)(r);
  float4 p1 = *(const float4*)(r + 4);
  float4 p2 = *(const float4*)(r + 16);
  float4 p3 = *(const float4*)(r + 20);
  union { v2bf h[8]; v16bf v; } x;
  x.h[0] = cvt2(p0.x, p0.y); x.h[1] = cvt2(p0.z, p0.w);
  x.h[2] = cvt2(p1.x, p1.y); x.h[3] = cvt2(p1.z, p1.w);
  x.h[4] = cvt2(p2.x, p2.y); x.h[5] = cvt2(p2.z, p2.w);
  x.h[6] = cvt2(p3.x, p3.y); x.h[7] = cvt2(p3.z, p3.w);
  return x.v;
}

// A fragment from LDS bf16 [16 x ld] block: two ds_load_b128.
__device__ __forceinline__ v16bf load_a_lds(const __bf16* base, int ld, int kc, int lane){
  const __bf16* r = base + (lane & 15) * ld + kc + ((lane >> 4) << 3);
  union { uint4 q[2]; v16bf v; } x;
  x.q[0] = *(const uint4*)(r);
  x.q[1] = *(const uint4*)(r + 16);
  return x.v;
}

// B fragment: two b128 loads of pre-packed bf16 data.
__device__ __forceinline__ v16bf load_b_frag(const uint32_t* fb, int lane){
  union { uint4 q[2]; v16bf v; } x;
  const uint4* p = (const uint4*)(fb + lane * 8);
  x.q[0] = p[0];
  x.q[1] = p[1];
  return x.v;
}

__device__ __forceinline__ v8f wmma_bf(v16bf a, v16bf b, v8f c){
  return __builtin_amdgcn_wmma_f32_16x16x32_bf16(false, a, false, b, (short)0, c, false, false);
}

// Row reductions within a 16-lane half-wave (wave32).
__device__ __forceinline__ float hsum16(float v){
  v += __shfl_xor(v, 1, 32); v += __shfl_xor(v, 2, 32);
  v += __shfl_xor(v, 4, 32); v += __shfl_xor(v, 8, 32);
  return v;
}
__device__ __forceinline__ float hmax16(float v){
  v = fmaxf(v, __shfl_xor(v, 1, 32)); v = fmaxf(v, __shfl_xor(v, 2, 32));
  v = fmaxf(v, __shfl_xor(v, 4, 32)); v = fmaxf(v, __shfl_xor(v, 8, 32));
  return v;
}
__device__ __forceinline__ float eluf(float v){ return v > 0.f ? v : __expf(v) - 1.f; }
__device__ __forceinline__ float sigmf(float v){ return 1.f / (1.f + __expf(-v)); }

#define LDS_FENCE() asm volatile("s_wait_dscnt 0" ::: "memory")

// ---------------------------------------------------------------------------
// Main kernel: 8 waves/block, each wave owns rows [b0, b0+16).
// ---------------------------------------------------------------------------
__global__ __launch_bounds__(256)
void svs_main_kernel(const float* __restrict__ x,
                     const float* __restrict__ mb1,  const float* __restrict__ mb2,
                     const float* __restrict__ mlb,  const float* __restrict__ mgb,
                     const float* __restrict__ mlb2, const float* __restrict__ mgamma,
                     const float* __restrict__ mbeta,
                     const float* __restrict__ b1,   const float* __restrict__ b2,
                     const float* __restrict__ gb,   const float* __restrict__ lb,
                     const float* __restrict__ gamma,const float* __restrict__ beta,
                     const uint32_t* __restrict__ ws,
                     float* __restrict__ out, float* __restrict__ maskout)
{
  __shared__ __bf16 sh_h [8][16 * 64];   // intermediate [16,64] bf16 per wave
  __shared__ __bf16 sh_h2[8][16 * 64];
  __shared__ __bf16 sh_hs[8][16 * 32];   // mask-branch h [16,32]
  __shared__ float  sh_mask[8][16 * 32]; // softmax mask [16,32]

  const int lane = threadIdx.x & 31;
  const int wid  = threadIdx.x >> 5;
  const int m    = lane & 15;
  const int hf   = lane >> 4;            // half-wave id: C/D row M = r + 8*hf
  const int b0   = (blockIdx.x * 8 + wid) * 16;
  const float* xt = x + (size_t)b0 * IN_DIM;

  __bf16* hb = sh_h[wid];
  __bf16* h2 = sh_h2[wid];
  __bf16* hs = sh_hs[wid];
  float*  mk = sh_mask[wid];

  // ===================== Phase A: StaticGRN mask branch =====================
  v8f acc1[4], accR[2];
#pragma unroll
  for (int t = 0; t < 4; ++t) acc1[t] = vzero8();
#pragma unroll
  for (int t = 0; t < 2; ++t) accR[t] = vzero8();

  // flat@mw1 ([16,2048]x[2048,64]) fused with flat@mlw ([2048,32])
  for (int c = 0; c < 64; ++c){
    v16bf a = load_a_global(xt, IN_DIM, c * 32, lane);
#pragma unroll
    for (int t = 0; t < 4; ++t)
      acc1[t] = wmma_bf(a, load_b_frag(ws + OFF_MW1 + (c * 4 + t) * 256, lane), acc1[t]);
#pragma unroll
    for (int t = 0; t < 2; ++t)
      accR[t] = wmma_bf(a, load_b_frag(ws + OFF_MLW + (c * 2 + t) * 256, lane), accR[t]);
  }

  // h1 = ELU(. + mb1) -> LDS bf16
#pragma unroll
  for (int t = 0; t < 4; ++t){
    float bias = mb1[t * 16 + m];
#pragma unroll
    for (int r = 0; r < 8; ++r)
      hb[(r + 8 * hf) * 64 + t * 16 + m] = tobf(eluf(acc1[t][r] + bias));
  }
  LDS_FENCE();

  // h = h1 @ mw2 + mb2  ([16,64]x[64,32])
  v8f accH[2];
#pragma unroll
  for (int t = 0; t < 2; ++t) accH[t] = vzero8();
#pragma unroll
  for (int c = 0; c < 2; ++c){
    v16bf a = load_a_lds(hb, 64, c * 32, lane);
#pragma unroll
    for (int t = 0; t < 2; ++t)
      accH[t] = wmma_bf(a, load_b_frag(ws + OFF_MW2 + (c * 2 + t) * 256, lane), accH[t]);
  }
#pragma unroll
  for (int t = 0; t < 2; ++t){
    float bias = mb2[t * 16 + m];
#pragma unroll
    for (int r = 0; r < 8; ++r)
      hs[(r + 8 * hf) * 32 + t * 16 + m] = tobf(accH[t][r] + bias);
  }
  LDS_FENCE();

  // gate & linear heads ([16,32]x[32,32] each)
  v8f accG[2], accL[2];
#pragma unroll
  for (int t = 0; t < 2; ++t){ accG[t] = vzero8(); accL[t] = vzero8(); }
  {
    v16bf a = load_a_lds(hs, 32, 0, lane);
#pragma unroll
    for (int t = 0; t < 2; ++t)
      accG[t] = wmma_bf(a, load_b_frag(ws + OFF_MGW + t * 256, lane), accG[t]);
#pragma unroll
    for (int t = 0; t < 2; ++t)
      accL[t] = wmma_bf(a, load_b_frag(ws + OFF_MLW2 + t * 256, lane), accL[t]);
  }

  // pre = sigmoid(g)*l + res ; then LN over F=32 + softmax -> mask
  float pre[2][8];
#pragma unroll
  for (int t = 0; t < 2; ++t){
    int N = t * 16 + m;
    float gbv = mgb[N], l2b = mlb2[N], rb = mlb[N];
#pragma unroll
    for (int r = 0; r < 8; ++r)
      pre[t][r] = sigmf(accG[t][r] + gbv) * (accL[t][r] + l2b) + accR[t][r] + rb;
  }

  float mg0 = mgamma[m], mg1 = mgamma[16 + m];
  float mt0 = mbeta[m],  mt1 = mbeta[16 + m];
#pragma unroll
  for (int r = 0; r < 8; ++r){
    float mean = hsum16(pre[0][r] + pre[1][r]) * (1.f / 32.f);
    float d0 = pre[0][r] - mean, d1 = pre[1][r] - mean;
    float var  = hsum16(d0 * d0 + d1 * d1) * (1.f / 32.f);
    float rstd = __frsqrt_rn(var + EPSL);
    float l0 = d0 * rstd * mg0 + mt0;
    float l1 = d1 * rstd * mg1 + mt1;
    float mx = hmax16(fmaxf(l0, l1));
    float e0 = __expf(l0 - mx), e1 = __expf(l1 - mx);
    float inv = 1.f / hsum16(e0 + e1);
    int M = r + 8 * hf;
    float m0 = e0 * inv, m1 = e1 * inv;
    mk[M * 32 + m]      = m0;
    mk[M * 32 + 16 + m] = m1;
    maskout[(size_t)(b0 + M) * 32 + m]      = m0;
    maskout[(size_t)(b0 + M) * 32 + 16 + m] = m1;
  }
  LDS_FENCE();

  // ===================== Phase B: per-feature GRNs =====================
  v8f oacc[4];
#pragma unroll
  for (int t = 0; t < 4; ++t) oacc[t] = vzero8();

  for (int f = 0; f < 32; ++f){
    const float* xf = xt + f * 64;

    // hh0 = ELU(x_f @ w1_f + b1_f)
    v8f acc[4];
#pragma unroll
    for (int t = 0; t < 4; ++t) acc[t] = vzero8();
#pragma unroll
    for (int c = 0; c < 2; ++c){
      v16bf a = load_a_global(xf, IN_DIM, c * 32, lane);
#pragma unroll
      for (int t = 0; t < 4; ++t)
        acc[t] = wmma_bf(a, load_b_frag(ws + OFF_W1 + ((f * 2 + c) * 4 + t) * 256, lane), acc[t]);
    }
#pragma unroll
    for (int t = 0; t < 4; ++t){
      float bias = b1[f * 64 + t * 16 + m];
#pragma unroll
      for (int r = 0; r < 8; ++r)
        hb[(r + 8 * hf) * 64 + t * 16 + m] = tobf(eluf(acc[t][r] + bias));
    }
    LDS_FENCE();

    // hh = hh0 @ w2_f + b2_f
#pragma unroll
    for (int t = 0; t < 4; ++t) acc[t] = vzero8();
#pragma unroll
    for (int c = 0; c < 2; ++c){
      v16bf a = load_a_lds(hb, 64, c * 32, lane);
#pragma unroll
      for (int t = 0; t < 4; ++t)
        acc[t] = wmma_bf(a, load_b_frag(ws + OFF_W2 + ((f * 2 + c) * 4 + t) * 256, lane), acc[t]);
    }
#pragma unroll
    for (int t = 0; t < 4; ++t){
      float bias = b2[f * 64 + t * 16 + m];
#pragma unroll
      for (int r = 0; r < 8; ++r)
        h2[(r + 8 * hf) * 64 + t * 16 + m] = tobf(acc[t][r] + bias);
    }
    LDS_FENCE();

    // gate / lin heads
    v8f ag[4], al[4];
#pragma unroll
    for (int t = 0; t < 4; ++t){ ag[t] = vzero8(); al[t] = vzero8(); }
#pragma unroll
    for (int c = 0; c < 2; ++c){
      v16bf a = load_a_lds(h2, 64, c * 32, lane);
#pragma unroll
      for (int t = 0; t < 4; ++t)
        ag[t] = wmma_bf(a, load_b_frag(ws + OFF_GW + ((f * 2 + c) * 4 + t) * 256, lane), ag[t]);
#pragma unroll
      for (int t = 0; t < 4; ++t)
        al[t] = wmma_bf(a, load_b_frag(ws + OFF_LW + ((f * 2 + c) * 4 + t) * 256, lane), al[t]);
    }

    // v = sigmoid(gate)*lin + x   (fp32 residual straight from global)
    float vv[4][8];
#pragma unroll
    for (int t = 0; t < 4; ++t){
      int N = t * 16 + m;
      float gbv = gb[f * 64 + N], lbv = lb[f * 64 + N];
#pragma unroll
      for (int r = 0; r < 8; ++r){
        int M = r + 8 * hf;
        float xe = xf[(size_t)M * IN_DIM + N];
        vv[t][r] = sigmf(ag[t][r] + gbv) * (al[t][r] + lbv) + xe;
      }
    }

    // LN over D=64, then weighted accumulate with mask[b,f]
    float mv[8];
#pragma unroll
    for (int r = 0; r < 8; ++r) mv[r] = mk[(r + 8 * hf) * 32 + f];

#pragma unroll
    for (int r = 0; r < 8; ++r){
      float s = vv[0][r] + vv[1][r] + vv[2][r] + vv[3][r];
      float mean = hsum16(s) * (1.f / 64.f);
      float q = 0.f;
#pragma unroll
      for (int t = 0; t < 4; ++t){ float d = vv[t][r] - mean; q += d * d; }
      float var  = hsum16(q) * (1.f / 64.f);
      float rstd = __frsqrt_rn(var + EPSL);
#pragma unroll
      for (int t = 0; t < 4; ++t){
        int N = t * 16 + m;
        float ln = (vv[t][r] - mean) * rstd * gamma[f * 64 + N] + beta[f * 64 + N];
        oacc[t][r] += mv[r] * ln;
      }
    }
  }

  // store out [B,64]
#pragma unroll
  for (int t = 0; t < 4; ++t)
#pragma unroll
    for (int r = 0; r < 8; ++r)
      out[(size_t)(b0 + r + 8 * hf) * 64 + t * 16 + m] = oacc[t][r];
}

// ---------------------------------------------------------------------------
extern "C" void kernel_launch(void* const* d_in, const int* in_sizes, int n_in,
                              void* d_out, int out_size, void* d_ws, size_t ws_size,
                              hipStream_t stream){
  const float* x     = (const float*)d_in[0];
  const float* mw1   = (const float*)d_in[1];
  const float* mb1   = (const float*)d_in[2];
  const float* mw2   = (const float*)d_in[3];
  const float* mb2   = (const float*)d_in[4];
  const float* mlw   = (const float*)d_in[5];
  const float* mlb   = (const float*)d_in[6];
  const float* mgw   = (const float*)d_in[7];
  const float* mgb   = (const float*)d_in[8];
  const float* mlw2  = (const float*)d_in[9];
  const float* mlb2  = (const float*)d_in[10];
  const float* mgamma= (const float*)d_in[11];
  const float* mbeta = (const float*)d_in[12];
  const float* w1    = (const float*)d_in[13];
  const float* b1    = (const float*)d_in[14];
  const float* w2    = (const float*)d_in[15];
  const float* b2    = (const float*)d_in[16];
  const float* gw    = (const float*)d_in[17];
  const float* gb    = (const float*)d_in[18];
  const float* lw    = (const float*)d_in[19];
  const float* lb    = (const float*)d_in[20];
  const float* gamma = (const float*)d_in[21];
  const float* beta  = (const float*)d_in[22];

  uint32_t* ws      = (uint32_t*)d_ws;
  float*    out     = (float*)d_out;
  float*    maskout = out + (size_t)B_ROWS * 64;

  // Pre-pack all weights into bf16 WMMA B-fragment layout (deterministic,
  // rebuilt every call; ~1.45 MB of d_ws used).
  pack_weights_kernel<<<256, 256, 0, stream>>>(mw1,  ws + OFF_MW1,  2048, 64, 1);
  pack_weights_kernel<<<128, 256, 0, stream>>>(mlw,  ws + OFF_MLW,  2048, 32, 1);
  pack_weights_kernel<<<4,   256, 0, stream>>>(mw2,  ws + OFF_MW2,  64,   32, 1);
  pack_weights_kernel<<<2,   256, 0, stream>>>(mgw,  ws + OFF_MGW,  32,   32, 1);
  pack_weights_kernel<<<2,   256, 0, stream>>>(mlw2, ws + OFF_MLW2, 32,   32, 1);
  pack_weights_kernel<<<256, 256, 0, stream>>>(w1,   ws + OFF_W1,   64,   64, 32);
  pack_weights_kernel<<<256, 256, 0, stream>>>(w2,   ws + OFF_W2,   64,   64, 32);
  pack_weights_kernel<<<256, 256, 0, stream>>>(gw,   ws + OFF_GW,   64,   64, 32);
  pack_weights_kernel<<<256, 256, 0, stream>>>(lw,   ws + OFF_LW,   64,   64, 32);

  // 16384 rows / (16 rows/wave * 8 waves/block) = 128 blocks
  svs_main_kernel<<<128, 256, 0, stream>>>(x, mb1, mb2, mlb, mgb, mlb2, mgamma, mbeta,
                                           b1, b2, gb, lb, gamma, beta,
                                           ws, out, maskout);
}
